// LSTMTagger_6433861009408
// MI455X (gfx1250) — compile-verified, matrix operand order checked by reading, not run on hardware
//
#include <hip/hip_runtime.h>
#include <hip/hip_bf16.h>
#include <math.h>

// ---------------- types for WMMA ----------------
typedef __attribute__((ext_vector_type(16))) _Float16 v16h;
typedef __attribute__((ext_vector_type(8)))  _Float16 v8h;
typedef __attribute__((ext_vector_type(8)))  float    v8f;

// Problem sizes (match reference)
#define S_LEN 1024
#define L_LEN 16
#define DC 128
#define HC 256
#define DW 512
#define HW 512
#define NTAGS 64
#define T_CHAR (S_LEN * L_LEN)   // 16384

// dynamic LDS budget for char scan: 256KB weight cache + h + g
#define CH_WCOLS 128                                   // K-cols of cW_hh cached in LDS
#define CH_SMEM  (1024 * CH_WCOLS * 2 + HC * 4 + 4 * HC * 4)  // 267264 B < 320KB WGP LDS

// ---------------- small utility kernels ----------------

__global__ __launch_bounds__(256) void f32_to_f16_kernel(const float* __restrict__ src,
                                                         _Float16* __restrict__ dst, int n) {
  int i = blockIdx.x * blockDim.x + threadIdx.x;
  if (i < n) dst[i] = (_Float16)src[i];
}

// Gather char embeddings -> f16 matrix Xc [T_CHAR, DC]
__global__ __launch_bounds__(256) void embed_chars_kernel(const int* __restrict__ chars,
                                                          const float* __restrict__ char_emb,
                                                          _Float16* __restrict__ Xc, int n) {
  int i = blockIdx.x * blockDim.x + threadIdx.x;
  if (i >= n) return;
  int pos = i >> 7;          // / DC
  int d   = i & (DC - 1);
  int row = chars[pos];
  Xc[i] = (_Float16)char_emb[row * DC + d];
}

// embeds[s] = concat(word_emb[sentence[s]] (512), char_last[s] (256)) -> f16 [S, 768]
__global__ __launch_bounds__(256) void build_embeds_kernel(const int* __restrict__ sentence,
                                                           const float* __restrict__ word_emb,
                                                           const float* __restrict__ char_last,
                                                           _Float16* __restrict__ embeds, int n) {
  int i = blockIdx.x * blockDim.x + threadIdx.x;
  if (i >= n) return;
  int s = i / (DW + HC);
  int d = i - s * (DW + HC);
  float v;
  if (d < DW) v = word_emb[(size_t)sentence[s] * DW + d];
  else        v = char_last[(size_t)s * HC + (d - DW)];
  embeds[i] = (_Float16)v;
}

// ---------------- WMMA GEMM: out[M,N] = A[M,K] @ W[N,K]^T + b0 + b1 ----------------
// One 16x16 output tile per wave; 8 waves per 256-thread block.
// Fragment layouts per CDNA5 ISA 7.12.2 (16-bit A 16x32; f32 C/D 16x16).
__global__ __launch_bounds__(256) void wmma_gemm_bias_kernel(
    const _Float16* __restrict__ A,   // [M,K] row-major
    const _Float16* __restrict__ W,   // [N,K] row-major (B = W^T)
    const float* __restrict__ b0,     // [N] or nullptr
    const float* __restrict__ b1,     // [N] or nullptr
    float* __restrict__ out,          // [M,N]
    int M, int N, int K) {
  int wave = (blockIdx.x * 256 + threadIdx.x) >> 5;
  int lane = threadIdx.x & 31;
  int tilesN = N >> 4;
  int tm = (wave / tilesN) << 4;
  int tn = (wave - (wave / tilesN) * tilesN) << 4;
  if (tm >= M) return;

  int half = lane >> 4;     // selects K-half per ISA layout
  int ln   = lane & 15;     // row (A) / col (B) within tile

  const _Float16* arow = A + (size_t)(tm + ln) * K + half * 8;
  const _Float16* wrow = W + (size_t)(tn + ln) * K + half * 8;

  union Frag { v16h v; v8h h[2]; };
  v8f acc = {};
  for (int k0 = 0; k0 < K; k0 += 32) {
    Frag a, b;
    a.h[0] = *(const v8h*)(arow + k0);        // K = k0 + half*8 .. +7
    a.h[1] = *(const v8h*)(arow + k0 + 16);   // K = k0 + 16 + half*8 .. +7
    b.h[0] = *(const v8h*)(wrow + k0);
    b.h[1] = *(const v8h*)(wrow + k0 + 16);
    acc = __builtin_amdgcn_wmma_f32_16x16x32_f16(
        /*neg_a=*/false, a.v, /*neg_b=*/false, b.v,
        /*c_mod=*/(short)0, acc, /*reuse_a=*/false, /*reuse_b=*/false);
  }

  int n = tn + ln;
  float bias = 0.0f;
  if (b0) bias += b0[n];
  if (b1) bias += b1[n];
  size_t base = (size_t)(tm + half * 8) * N + n;
#pragma unroll
  for (int r = 0; r < 8; ++r) out[base + (size_t)r * N] = acc[r] + bias;
}

// 8-wide f16*f32 dot helper (weights f16, h f32)
__device__ __forceinline__ float dot8h(const v8h w, const float* __restrict__ hb) {
  float s = 0.0f;
#pragma unroll
  for (int i = 0; i < 8; ++i) s += (float)w[i] * hb[i];
  return s;
}

// ---------------- char LSTM scan (sequential over 16384 steps) ----------------
// One workgroup (32 wave32s) on one WGP. h in LDS; first 128 K-cols of cW_hh
// (f16, 256KB) pinned in the 320KB WGP LDS, remaining 128 cols streamed from L2.
__global__ __launch_bounds__(1024) void char_scan_kernel(
    const float* __restrict__ xg,          // [T_CHAR, 4*HC]
    const _Float16* __restrict__ Whh,      // [4*HC, HC] f16
    float* __restrict__ char_last) {       // [S, HC]
  extern __shared__ char smem[];
  _Float16* wl = (_Float16*)smem;                          // [1024][CH_WCOLS]
  float* h = (float*)(smem + 1024 * CH_WCOLS * 2);         // [HC]
  float* g = h + HC;                                       // [4*HC]

  int j = threadIdx.x;  // 0..1023, one gate row per thread
  // cooperative weight-cache fill: thread j copies its own row's low K-half
  {
    const v8h* src = (const v8h*)(Whh + (size_t)j * HC);
    v8h* dst = (v8h*)(wl + (size_t)j * CH_WCOLS);
#pragma unroll
    for (int k8 = 0; k8 < CH_WCOLS / 8; ++k8) dst[k8] = src[k8];
  }
  if (j < HC) h[j] = 0.0f;
  float c = 0.0f;
  __syncthreads();

  const v8h* wlo = (const v8h*)(wl + (size_t)j * CH_WCOLS);         // LDS-cached K 0..127
  const v8h* whi = (const v8h*)(Whh + (size_t)j * HC + CH_WCOLS);   // streamed   K 128..255
  for (int t = 0; t < T_CHAR; ++t) {
    // prefetch next step's gate pre-activation while we compute this one
    if (t + 1 < T_CHAR) __builtin_prefetch(xg + (size_t)(t + 1) * (4 * HC) + j, 0, 0);
    float acc = xg[(size_t)t * (4 * HC) + j];
#pragma unroll 4
    for (int k8 = 0; k8 < CH_WCOLS / 8; ++k8) acc += dot8h(wlo[k8], h + 8 * k8);
#pragma unroll 4
    for (int k8 = 0; k8 < (HC - CH_WCOLS) / 8; ++k8)
      acc += dot8h(whi[k8], h + CH_WCOLS + 8 * k8);
    g[j] = acc;
    __syncthreads();
    if (j < HC) {
      float iv = g[j], fv = g[j + HC], gv = g[j + 2 * HC], ov = g[j + 3 * HC];
      float si = 1.0f / (1.0f + expf(-iv));
      float sf = 1.0f / (1.0f + expf(-fv));
      float so = 1.0f / (1.0f + expf(-ov));
      c = sf * c + si * tanhf(gv);
      float hn = so * tanhf(c);
      h[j] = hn;
      if ((t & (L_LEN - 1)) == (L_LEN - 1)) char_last[(size_t)(t >> 4) * HC + j] = hn;
    }
    __syncthreads();
  }
}

// ---------------- word LSTM scan (1024 steps, f16 streamed weights) ----------------
__global__ __launch_bounds__(1024) void word_scan_kernel(
    const float* __restrict__ xg,          // [S, 4*HW]
    const _Float16* __restrict__ Whh,      // [4*HW, HW] f16
    float* __restrict__ wh) {              // [S, HW]
  __shared__ float h[HW];
  __shared__ float g[4 * HW];
  int j = threadIdx.x;  // handles gate rows j and j+1024
  if (j < HW) h[j] = 0.0f;
  float c = 0.0f;
  __syncthreads();
  const v8h* w0 = (const v8h*)(Whh + (size_t)j * HW);
  const v8h* w1 = (const v8h*)(Whh + (size_t)(j + 1024) * HW);
  for (int t = 0; t < S_LEN; ++t) {
    if (t + 1 < S_LEN) __builtin_prefetch(xg + (size_t)(t + 1) * (4 * HW) + j, 0, 0);
    float a0 = xg[(size_t)t * (4 * HW) + j];
    float a1 = xg[(size_t)t * (4 * HW) + j + 1024];
#pragma unroll 4
    for (int k8 = 0; k8 < HW / 8; ++k8) {
      const float* hb = h + 8 * k8;
      a0 += dot8h(w0[k8], hb);
      a1 += dot8h(w1[k8], hb);
    }
    g[j] = a0;
    g[j + 1024] = a1;
    __syncthreads();
    if (j < HW) {
      float iv = g[j], fv = g[j + HW], gv = g[j + 2 * HW], ov = g[j + 3 * HW];
      float si = 1.0f / (1.0f + expf(-iv));
      float sf = 1.0f / (1.0f + expf(-fv));
      float so = 1.0f / (1.0f + expf(-ov));
      c = sf * c + si * tanhf(gv);
      float hn = so * tanhf(c);
      h[j] = hn;
      wh[(size_t)t * HW + j] = hn;
    }
    __syncthreads();
  }
}

// ---------------- log_softmax over axis 0 (per tag column) ----------------
__global__ __launch_bounds__(256) void logsoftmax_col_kernel(const float* __restrict__ tag,
                                                             float* __restrict__ out) {
  int col = blockIdx.x;      // 0..63
  int tid = threadIdx.x;
  __shared__ float red[256];
  float m = -3.4e38f;
  for (int s = tid; s < S_LEN; s += 256) m = fmaxf(m, tag[(size_t)s * NTAGS + col]);
  red[tid] = m;
  __syncthreads();
  for (int w = 128; w > 0; w >>= 1) {
    if (tid < w) red[tid] = fmaxf(red[tid], red[tid + w]);
    __syncthreads();
  }
  float M = red[0];
  __syncthreads();
  float sum = 0.0f;
  for (int s = tid; s < S_LEN; s += 256) sum += expf(tag[(size_t)s * NTAGS + col] - M);
  red[tid] = sum;
  __syncthreads();
  for (int w = 128; w > 0; w >>= 1) {
    if (tid < w) red[tid] += red[tid + w];
    __syncthreads();
  }
  float lse = M + logf(red[0]);
  for (int s = tid; s < S_LEN; s += 256)
    out[(size_t)s * NTAGS + col] = tag[(size_t)s * NTAGS + col] - lse;
}

// ---------------- launcher ----------------
extern "C" void kernel_launch(void* const* d_in, const int* in_sizes, int n_in,
                              void* d_out, int out_size, void* d_ws, size_t ws_size,
                              hipStream_t stream) {
  const int*   sentence = (const int*)  d_in[0];
  const int*   chars    = (const int*)  d_in[1];
  const float* char_emb = (const float*)d_in[2];
  const float* word_emb = (const float*)d_in[3];
  const float* cW_ih    = (const float*)d_in[4];   // [1024,128]
  const float* cW_hh    = (const float*)d_in[5];   // [1024,256]
  const float* cb_ih    = (const float*)d_in[6];
  const float* cb_hh    = (const float*)d_in[7];
  const float* wW_ih    = (const float*)d_in[8];   // [2048,768]
  const float* wW_hh    = (const float*)d_in[9];   // [2048,512]
  const float* wb_ih    = (const float*)d_in[10];
  const float* wb_hh    = (const float*)d_in[11];
  const float* W_tag    = (const float*)d_in[12];  // [64,512]
  const float* b_tag    = (const float*)d_in[13];
  float* out            = (float*)d_out;           // [1024,64]

  // workspace carve-out (256B aligned, every buffer written before read)
  char* ws = (char*)d_ws;
  size_t off = 0;
  auto carve = [&](size_t bytes) -> char* {
    char* p = ws + off;
    off = (off + bytes + 255) & ~(size_t)255;
    return p;
  };
  _Float16* Xc      = (_Float16*)carve((size_t)T_CHAR * DC * 2);        // 4 MiB
  _Float16* cWih_h  = (_Float16*)carve((size_t)4 * HC * DC * 2);        // 256 KiB
  _Float16* cWhh_h  = (_Float16*)carve((size_t)4 * HC * HC * 2);        // 512 KiB
  _Float16* wWih_h  = (_Float16*)carve((size_t)4 * HW * (DW + HC) * 2); // 3 MiB
  _Float16* wWhh_h  = (_Float16*)carve((size_t)4 * HW * HW * 2);        // 2 MiB
  _Float16* emb_h   = (_Float16*)carve((size_t)S_LEN * (DW + HC) * 2);  // 1.5 MiB
  _Float16* wh_h    = (_Float16*)carve((size_t)S_LEN * HW * 2);         // 1 MiB
  _Float16* Wtag_h  = (_Float16*)carve((size_t)NTAGS * HW * 2);         // 64 KiB
  float*    clast   = (float*)   carve((size_t)S_LEN * HC * 4);         // 1 MiB
  float*    xg_w    = (float*)   carve((size_t)S_LEN * 4 * HW * 4);     // 8 MiB
  float*    wh      = (float*)   carve((size_t)S_LEN * HW * 4);         // 2 MiB
  float*    tagsp   = (float*)   carve((size_t)S_LEN * NTAGS * 4);      // 256 KiB
  float*    xg_c    = (float*)   carve((size_t)T_CHAR * 4 * HC * 4);    // 64 MiB

  // 1) weight conversions to f16 (GEMM operands + recurrent weights)
  {
    int n = 4 * HC * DC;
    f32_to_f16_kernel<<<(n + 255) / 256, 256, 0, stream>>>(cW_ih, cWih_h, n);
  }
  {
    int n = 4 * HC * HC;
    f32_to_f16_kernel<<<(n + 255) / 256, 256, 0, stream>>>(cW_hh, cWhh_h, n);
  }
  {
    int n = 4 * HW * (DW + HC);
    f32_to_f16_kernel<<<(n + 255) / 256, 256, 0, stream>>>(wW_ih, wWih_h, n);
  }
  {
    int n = 4 * HW * HW;
    f32_to_f16_kernel<<<(n + 255) / 256, 256, 0, stream>>>(wW_hh, wWhh_h, n);
  }
  {
    int n = NTAGS * HW;
    f32_to_f16_kernel<<<(n + 255) / 256, 256, 0, stream>>>(W_tag, Wtag_h, n);
  }

  // 2) char embedding gather (f16)
  {
    int n = T_CHAR * DC;
    embed_chars_kernel<<<(n + 255) / 256, 256, 0, stream>>>(chars, char_emb, Xc, n);
  }

  // 3) char input projection: xg_c = Xc @ cW_ih^T + cb_ih + cb_hh   (WMMA)
  {
    int M = T_CHAR, N = 4 * HC, K = DC;
    int tiles = (M / 16) * (N / 16);               // 65536 tiles, 8 waves/block
    wmma_gemm_bias_kernel<<<tiles / 8, 256, 0, stream>>>(Xc, cWih_h, cb_ih, cb_hh,
                                                         xg_c, M, N, K);
  }

  // 4) sequential char LSTM scan (256KB of W_hh pinned in WGP LDS)
  char_scan_kernel<<<1, 1024, CH_SMEM, stream>>>(xg_c, cWhh_h, clast);

  // 5) build word-level inputs [we | char_last] (f16)
  {
    int n = S_LEN * (DW + HC);
    build_embeds_kernel<<<(n + 255) / 256, 256, 0, stream>>>(sentence, word_emb, clast, emb_h, n);
  }

  // 6) word input projection: xg_w = embeds @ wW_ih^T + wb_ih + wb_hh  (WMMA)
  {
    int M = S_LEN, N = 4 * HW, K = DW + HC;
    int tiles = (M / 16) * (N / 16);               // 8192 tiles
    wmma_gemm_bias_kernel<<<tiles / 8, 256, 0, stream>>>(emb_h, wWih_h, wb_ih, wb_hh,
                                                         xg_w, M, N, K);
  }

  // 7) sequential word LSTM scan -> wh [S, HW]
  word_scan_kernel<<<1, 1024, 0, stream>>>(xg_w, wWhh_h, wh);

  // 8) tag projection: tagsp = wh @ W_tag^T + b_tag   (WMMA, wh -> f16)
  {
    int n = S_LEN * HW;
    f32_to_f16_kernel<<<(n + 255) / 256, 256, 0, stream>>>(wh, wh_h, n);
    int M = S_LEN, N = NTAGS, K = HW;
    int tiles = (M / 16) * (N / 16);               // 256 tiles
    wmma_gemm_bias_kernel<<<tiles / 8, 256, 0, stream>>>(wh_h, Wtag_h, b_tag, nullptr,
                                                         tagsp, M, N, K);
  }

  // 9) log_softmax over axis 0 (per tag column), write final output
  logsoftmax_col_kernel<<<NTAGS, 256, 0, stream>>>(tagsp, out);
}
